// LLC_12171937317104
// MI455X (gfx1250) — matrix operand order repaired and checked by reading, not compiled, not run on hardware
//
#include <hip/hip_runtime.h>
#include <hip/hip_bf16.h>

// ---------------- problem constants ----------------
#define TT   512            // rollout length
#define HB   256            // batch
#define NN   (TT * HB)      // 131072 rows
#define HID  128            // lstm hidden
#define GATE 512            // 4*HID
#define INW  560            // lstm input width
#define XK   576            // padded x width (multiple of 32)
#define K1   160            // padded prop layer-1 K (140 -> 160)

typedef __attribute__((ext_vector_type(16))) _Float16 v16h;
typedef __attribute__((ext_vector_type(8)))  _Float16 v8h;
typedef __attribute__((ext_vector_type(8)))  float    v8f;

// ---------------- fast activations (hardware TRANS ops on gfx1250) ----------------
__device__ static inline float fast_tanh(float x) {
#if __has_builtin(__builtin_amdgcn_tanhf)
    return __builtin_amdgcn_tanhf(x);           // v_tanh_f32
#else
    return tanhf(x);
#endif
}
__device__ static inline float fast_sigmoid(float x) {
    // 1/(1+exp(-x)) = rcp(1 + exp2(-x*log2(e)))
    float e = __builtin_amdgcn_exp2f(-x * 1.44269504088896340736f);   // v_exp_f32
    return __builtin_amdgcn_rcpf(1.0f + e);                           // v_rcp_f32
}

// ---------------- WMMA fragment helpers ----------------
// A fragment (16x32 f16), tile row-major [16][ld] in LDS.
__device__ static inline v16h load_a_frag_lds(const _Float16* tile, int ld, int k0, int lane) {
    int hi16 = (lane >> 4) & 1;
    int m = lane & 15;
    const _Float16* row = tile + m * ld + k0;
    v8h lo = *(const v8h*)(row + (hi16 ? 16 : 0));
    v8h hi = *(const v8h*)(row + (hi16 ? 24 : 8));
    v16h a;
#pragma unroll
    for (int i = 0; i < 8; ++i) { a[i] = lo[i]; a[8 + i] = hi[i]; }
    return a;
}

// B fragment (32x16, B = W^T), W row-major f16 [ncols][ldk]
__device__ static inline v16h load_b_frag(const _Float16* W, int ldk, int n0, int k0, int lane) {
    int n = n0 + (lane & 15);
    int koff = k0 + (((lane >> 4) & 1) ? 16 : 0);
    return *(const v16h*)(W + (size_t)n * ldk + koff);
}

// C/D fragment: lane = n + (m>=8)*16, vgpr r = m&7
__device__ static inline void store_c_frag_f32(float* out, int ldo, int n0, const v8f& acc, int lane) {
    int n = n0 + (lane & 15);
    int mb = ((lane >> 4) & 1) ? 8 : 0;
#pragma unroll
    for (int r = 0; r < 8; ++r) out[(size_t)(mb + r) * ldo + n] = acc[r];
}

__device__ static inline void store_c_frag_f16_lds(_Float16* tile, int ld, int n0, const v8f& acc, int lane) {
    int n = n0 + (lane & 15);
    int mb = ((lane >> 4) & 1) ? 8 : 0;
#pragma unroll
    for (int r = 0; r < 8; ++r) tile[(mb + r) * ld + n] = (_Float16)acc[r];
}

#define WMMA_F16(a, b, c) \
    __builtin_amdgcn_wmma_f32_16x16x32_f16(false, (a), false, (b), (short)0, (c), false, false)

// ---------------- kernel 0: weight prep (f32 -> f16, pad K) ----------------
__global__ __launch_bounds__(256) void prep_kernel(
    const float* W1, const float* W2, const float* W3, const float* Wpa,
    const float* Wih, const float* Whh, const float* Wc,
    const float* bih, const float* bhh,
    _Float16* W1h, _Float16* W2h, _Float16* W3h, _Float16* Wpah,
    _Float16* Wihh, _Float16* Whhh, _Float16* Wch, float* bsum)
{
    int stride = gridDim.x * blockDim.x;
    int t0 = blockIdx.x * blockDim.x + threadIdx.x;
    for (int p = t0; p < 128 * K1; p += stride) {
        int r = p / K1, k = p % K1;
        W1h[p] = (_Float16)(k < 140 ? W1[r * 140 + k] : 0.0f);
    }
    for (int p = t0; p < 128 * 128; p += stride) W2h[p] = (_Float16)W2[p];
    for (int p = t0; p < 128 * 128; p += stride) W3h[p] = (_Float16)W3[p];
    for (int p = t0; p < 32 * 32;  p += stride) Wpah[p] = (_Float16)Wpa[p];
    for (int p = t0; p < GATE * XK; p += stride) {
        int r = p / XK, k = p % XK;
        Wihh[p] = (_Float16)(k < INW ? Wih[r * INW + k] : 0.0f);
    }
    for (int p = t0; p < GATE * HID; p += stride) Whhh[p] = (_Float16)Whh[p];
    for (int p = t0; p < 128 * 128; p += stride) Wch[p] = (_Float16)Wc[p];
    for (int p = t0; p < GATE;      p += stride) bsum[p] = bih[p] + bhh[p];
}

// ---------------- kernel 1: fused MLP -> x f16 [N][576] ----------------
__global__ __launch_bounds__(256) void mlp_kernel(
    const float* obs, const float* hlc,
    const _Float16* W1h, const float* b1,
    const _Float16* W2h, const float* b2,
    const _Float16* W3h, const float* b3,
    const _Float16* Wpah, const float* bpa,
    _Float16* x)
{
    __shared__ __align__(16) _Float16 bufA[16 * K1];
    __shared__ __align__(16) _Float16 bufB[16 * 128];
    __shared__ __align__(16) _Float16 pa_in[16 * 32];
    __shared__ __align__(16) _Float16 pa_out[16 * 32];
    const int n0 = blockIdx.x * 16;
    const int tid = threadIdx.x, lane = tid & 31, wave = tid >> 5;

    for (int p = tid; p < 16 * K1; p += 256) {
        int r = p / K1, k = p % K1;
        float v = (k < 140) ? obs[(size_t)(n0 + r) * 181 + 9 + k] : 0.0f;
        bufA[p] = (_Float16)v;
    }
    for (int p = tid; p < 16 * 32; p += 256) {
        int r = p >> 5, k = p & 31;
        pa_in[p] = (_Float16)obs[(size_t)(n0 + r) * 181 + 149 + k];
    }
    __syncthreads();

    const int nc = wave * 16;
    {   // layer 1: K=160
        v8f acc = {};
#pragma unroll
        for (int kc = 0; kc < K1; kc += 32) {
            v16h a = load_a_frag_lds(bufA, K1, kc, lane);
            v16h b = load_b_frag(W1h, K1, nc, kc, lane);
            acc = WMMA_F16(a, b, acc);
        }
        float bb = b1[nc + (lane & 15)];
#pragma unroll
        for (int r = 0; r < 8; ++r) acc[r] = fast_tanh(acc[r] + bb);
        store_c_frag_f16_lds(bufB, 128, nc, acc, lane);
    }
    if (wave < 2) {  // prev-action layer: 32 outputs, K=32
        v16h a = load_a_frag_lds(pa_in, 32, 0, lane);
        v16h b = load_b_frag(Wpah, 32, nc, 0, lane);
        v8f acc = {};
        acc = WMMA_F16(a, b, acc);
        float bb = bpa[nc + (lane & 15)];
#pragma unroll
        for (int r = 0; r < 8; ++r) acc[r] = fast_tanh(acc[r] + bb);
        store_c_frag_f16_lds(pa_out, 32, nc, acc, lane);
    }
    __syncthreads();

    {   // layer 2: bufB -> bufA
        v8f acc = {};
#pragma unroll
        for (int kc = 0; kc < 128; kc += 32) {
            v16h a = load_a_frag_lds(bufB, 128, kc, lane);
            v16h b = load_b_frag(W2h, 128, nc, kc, lane);
            acc = WMMA_F16(a, b, acc);
        }
        float bb = b2[nc + (lane & 15)];
#pragma unroll
        for (int r = 0; r < 8; ++r) acc[r] = fast_tanh(acc[r] + bb);
        store_c_frag_f16_lds(bufA, 128, nc, acc, lane);
    }
    __syncthreads();
    {   // layer 3: bufA -> bufB
        v8f acc = {};
#pragma unroll
        for (int kc = 0; kc < 128; kc += 32) {
            v16h a = load_a_frag_lds(bufA, 128, kc, lane);
            v16h b = load_b_frag(W3h, 128, nc, kc, lane);
            acc = WMMA_F16(a, b, acc);
        }
        float bb = b3[nc + (lane & 15)];
#pragma unroll
        for (int r = 0; r < 8; ++r) acc[r] = fast_tanh(acc[r] + bb);
        store_c_frag_f16_lds(bufB, 128, nc, acc, lane);
    }
    __syncthreads();

    // assemble x row tile: [prop(128) | pa(32) | hlc(400) | pad(16)]
    for (int p = tid; p < 16 * XK; p += 256) {
        int r = p / XK, k = p % XK;
        _Float16 v;
        if (k < 128)      v = bufB[r * 128 + k];
        else if (k < 160) v = pa_out[r * 32 + (k - 128)];
        else if (k < 560) v = (_Float16)hlc[(size_t)(n0 + r) * 400 + (k - 160)];
        else              v = (_Float16)0.0f;
        x[(size_t)(n0 + r) * XK + k] = v;
    }
}

// ---------------- kernel 2: xg = x @ W_ih^T + (b_ih+b_hh), f32 [N][512] ----------------
// 32 output rows per block (2x B-fragment reuse); x tile staged via CDNA5
// async global->LDS copies (ASYNCcnt).
__global__ __launch_bounds__(256) void xg_gemm(
    const _Float16* x, const _Float16* Wihh, const float* bsum, float* xg)
{
    __shared__ __align__(16) _Float16 at[32 * XK];   // 36 KB
    const int n0 = blockIdx.x * 32;
    const int tid = threadIdx.x, lane = tid & 31, wave = tid >> 5;

    // async stage: 32*576 halves = 2304 x 16B chunks, 9 per thread
    {
        const char* src = (const char*)(x + (size_t)n0 * XK);
        unsigned int ldsbase = (unsigned int)(uintptr_t)(&at[0]);
#pragma unroll
        for (int i = 0; i < 9; ++i) {
            int p = tid + 256 * i;
            unsigned long long ga = (unsigned long long)(uintptr_t)(src) + (unsigned long long)p * 16u;
            unsigned int la = ldsbase + (unsigned int)p * 16u;
            asm volatile("global_load_async_to_lds_b128 %0, %1, off"
                         :: "v"(la), "v"(ga) : "memory");
        }
        asm volatile("s_wait_asynccnt 0x0" ::: "memory");
    }
    __syncthreads();

    const int col0 = wave * 64;
    v8f acc0[4], acc1[4];
#pragma unroll
    for (int j = 0; j < 4; ++j) {
        float bb = bsum[col0 + 16 * j + (lane & 15)];
#pragma unroll
        for (int r = 0; r < 8; ++r) { acc0[j][r] = bb; acc1[j][r] = bb; }
    }
#pragma unroll 2
    for (int kc = 0; kc < XK; kc += 32) {
        v16h a0 = load_a_frag_lds(at, XK, kc, lane);
        v16h a1 = load_a_frag_lds(at + 16 * XK, XK, kc, lane);
#pragma unroll
        for (int j = 0; j < 4; ++j) {
            v16h b = load_b_frag(Wihh, XK, col0 + 16 * j, kc, lane);
            acc0[j] = WMMA_F16(a0, b, acc0[j]);
            acc1[j] = WMMA_F16(a1, b, acc1[j]);
        }
    }
    float* out0 = xg + (size_t)n0 * GATE;
    float* out1 = out0 + (size_t)16 * GATE;
#pragma unroll
    for (int j = 0; j < 4; ++j) {
        store_c_frag_f32(out0, GATE, col0 + 16 * j, acc0[j], lane);
        store_c_frag_f32(out1, GATE, col0 + 16 * j, acc1[j], lane);
    }
}

// ---------------- kernel 3: persistent LSTM scan ----------------
// 16 WGs x 512 threads (16 waves). Each WG owns 16 batch rows for all 512
// steps (recurrence is independent per batch row -> no inter-WG sync).
// Each wave covers 2 gate col-tiles; W_hh B-fragments live in VGPRs.
__global__ __launch_bounds__(512) void lstm_scan(
    const _Float16* Whhh, const float* xg, const float* done,
    const float* h0, const float* c0,
    float* hid, float* hT, float* cT)
{
    __shared__ __align__(16) _Float16 hs[16 * HID];   // 4 KB
    __shared__ __align__(16) float    cs[16 * HID];   // 8 KB
    __shared__ __align__(16) float    gs[16 * GATE];  // 32 KB activated gates
    __shared__ float mrow[16];
    const int b0 = blockIdx.x * 16;
    const int tid = threadIdx.x, lane = tid & 31, wave = tid >> 5;
    const int col0 = wave * 32;          // 2 col tiles per wave
    const int gt = wave >> 2;            // 0:i 1:f 2:g 3:o
    const int nlo = lane & 15;
    const int mb  = ((lane >> 4) & 1) ? 8 : 0;

    // W_hh B-fragments resident in VGPRs for all 512 steps
    v16h bf[2][4];
#pragma unroll
    for (int j = 0; j < 2; ++j)
#pragma unroll
        for (int kc = 0; kc < 4; ++kc)
            bf[j][kc] = load_b_frag(Whhh, HID, col0 + 16 * j, kc * 32, lane);

    for (int p = tid; p < 16 * HID; p += 512) {
        int r = p >> 7, k = p & 127;
        hs[p] = (_Float16)h0[(size_t)(b0 + r) * HID + k];
        cs[p] = c0[(size_t)(b0 + r) * HID + k];
    }
    __syncthreads();

    for (int t = 0; t < TT; ++t) {
        if (tid < 16) mrow[tid] = 1.0f - done[(size_t)t * HB + b0 + tid];
        __syncthreads();
        for (int p = tid; p < 16 * HID; p += 512) {
            float m = mrow[p >> 7];
            hs[p] = (_Float16)((float)hs[p] * m);
            cs[p] = cs[p] * m;
        }
        __syncthreads();

        v16h afr[4];
#pragma unroll
        for (int kc = 0; kc < 4; ++kc) afr[kc] = load_a_frag_lds(hs, HID, kc * 32, lane);

        const float* xgrow = xg + ((size_t)t * HB + b0) * GATE;
#pragma unroll
        for (int j = 0; j < 2; ++j) {
            int n = col0 + 16 * j + nlo;
            v8f acc;
#pragma unroll
            for (int r = 0; r < 8; ++r) acc[r] = xgrow[(size_t)(mb + r) * GATE + n];
#pragma unroll
            for (int kc = 0; kc < 4; ++kc) acc = WMMA_F16(afr[kc], bf[j][kc], acc);
#pragma unroll
            for (int r = 0; r < 8; ++r) {
                float v = acc[r];
                v = (gt == 2) ? fast_tanh(v) : fast_sigmoid(v);
                gs[(mb + r) * GATE + n] = v;
            }
        }
        __syncthreads();

        float* hout = hid + ((size_t)t * HB + b0) * HID;
#pragma unroll
        for (int e = 0; e < 4; ++e) {
            int p = tid + 512 * e;           // 0..2047 over [16 rows][128 cols]
            int r = p >> 7, k = p & 127;
            float gi = gs[r * GATE + k];
            float gf = gs[r * GATE + 128 + k];
            float gg = gs[r * GATE + 256 + k];
            float go = gs[r * GATE + 384 + k];
            float c = gf * cs[p] + gi * gg;
            float h = go * fast_tanh(c);
            cs[p] = c;
            hs[p] = (_Float16)h;
            hout[(size_t)r * HID + k] = h;
            if (t == TT - 1) {
                hT[(size_t)(b0 + r) * HID + k] = h;
                cT[(size_t)(b0 + r) * HID + k] = c;
            }
        }
        __syncthreads();
    }
}

// ---------------- kernel 4: in-place combine: hid = relu(hid @ Wc^T + bc) ----------
__global__ __launch_bounds__(256) void combine_kernel(
    const _Float16* Wch, const float* bc, float* hid)
{
    __shared__ __align__(16) _Float16 at[16 * HID];
    const int n0 = blockIdx.x * 16;
    const int tid = threadIdx.x, lane = tid & 31, wave = tid >> 5;
    float* base = hid + (size_t)n0 * HID;
    for (int p = tid; p < 16 * HID; p += 256) at[p] = (_Float16)base[p];
    __syncthreads();

    const int col0 = wave * 16;
    v8f acc = {};
#pragma unroll
    for (int kc = 0; kc < 4; ++kc) {
        v16h a = load_a_frag_lds(at, HID, kc * 32, lane);
        v16h b = load_b_frag(Wch, HID, col0, kc * 32, lane);
        acc = WMMA_F16(a, b, acc);
    }
    const int n = col0 + (lane & 15);
    const int mb = ((lane >> 4) & 1) ? 8 : 0;
    float bb = bc[n];
#pragma unroll
    for (int r = 0; r < 8; ++r) {
        float v = acc[r] + bb;
        base[(size_t)(mb + r) * HID + n] = v > 0.0f ? v : 0.0f;
    }
}

// ---------------- host launcher ----------------
extern "C" void kernel_launch(void* const* d_in, const int* in_sizes, int n_in,
                              void* d_out, int out_size, void* d_ws, size_t ws_size,
                              hipStream_t stream) {
    const float* obs   = (const float*)d_in[0];
    const float* done  = (const float*)d_in[1];
    const float* hlc   = (const float*)d_in[2];
    const float* h0    = (const float*)d_in[3];
    const float* c0    = (const float*)d_in[4];
    const float* W1    = (const float*)d_in[5];
    const float* b1    = (const float*)d_in[6];
    const float* W2    = (const float*)d_in[7];
    const float* b2    = (const float*)d_in[8];
    const float* W3    = (const float*)d_in[9];
    const float* b3    = (const float*)d_in[10];
    const float* Wpa   = (const float*)d_in[11];
    const float* bpa   = (const float*)d_in[12];
    const float* Wih   = (const float*)d_in[13];
    const float* Whh   = (const float*)d_in[14];
    const float* bih   = (const float*)d_in[15];
    const float* bhh   = (const float*)d_in[16];
    const float* Wc    = (const float*)d_in[17];
    const float* bc    = (const float*)d_in[18];

    char* ws = (char*)d_ws;
    size_t off = 0;
    auto take = [&](size_t bytes) { char* p = ws + off; off = (off + bytes + 255) & ~(size_t)255; return p; };
    _Float16* Wihh = (_Float16*)take((size_t)GATE * XK * 2);
    _Float16* Whhh = (_Float16*)take((size_t)GATE * HID * 2);
    _Float16* W1h  = (_Float16*)take((size_t)128 * K1 * 2);
    _Float16* W2h  = (_Float16*)take((size_t)128 * 128 * 2);
    _Float16* W3h  = (_Float16*)take((size_t)128 * 128 * 2);
    _Float16* Wpah = (_Float16*)take((size_t)32 * 32 * 2);
    _Float16* Wch  = (_Float16*)take((size_t)128 * 128 * 2);
    float*    bsum = (float*)take((size_t)GATE * 4);
    _Float16* x    = (_Float16*)take((size_t)NN * XK * 2);
    float*    xg   = (float*)take((size_t)NN * GATE * 4);

    float* out = (float*)d_out;
    float* hid = out;                                   // [N,128]
    float* hT  = out + (size_t)NN * HID;                // [B,128]
    float* cT  = hT + (size_t)HB * HID;                 // [B,128]

    prep_kernel<<<512, 256, 0, stream>>>(W1, W2, W3, Wpa, Wih, Whh, Wc, bih, bhh,
                                         W1h, W2h, W3h, Wpah, Wihh, Whhh, Wch, bsum);
    mlp_kernel<<<NN / 16, 256, 0, stream>>>(obs, hlc, W1h, b1, W2h, b2, W3h, b3, Wpah, bpa, x);
    xg_gemm<<<NN / 32, 256, 0, stream>>>(x, Wihh, bsum, xg);
    lstm_scan<<<HB / 16, 512, 0, stream>>>(Whhh, xg, done, h0, c0, hid, hT, cT);
    combine_kernel<<<NN / 16, 256, 0, stream>>>(Wch, bc, hid);
}
